// LinearBlock_24936580121058
// MI455X (gfx1250) — matrix operand order repaired and checked by reading, not compiled, not run on hardware
//
#include <hip/hip_runtime.h>

// ---------------------------------------------------------------------------
// y = clip(x @ W^T, 0, 6),  x:[N,IN] f32,  W[o,i] = w_blocks[o>>3][i>>3][o&7][i&7]
// MI455X (gfx1250) plan:
//   pass 1: x f32 -> bf16 copy in scratch (done once; x reused 16x by GEMM)
//   pass 2: assemble W blocks -> dense bf16 Wd[out][in] (2 MB, L2-resident)
//   pass 3: GEMM: TDM tensor_load_to_lds (double-buffered, TENSORcnt waits)
//           -> ds_load_b128 fragments -> v_wmma_f32_16x16x32_bf16 -> ReLU6
// Roofline: 34.4 GFLOP; ~200 MB total HBM traffic => ~8.6 us at 23.3 TB/s,
// inner loop is pure TDM + LDS + WMMA so the bf16 tensor pipe (K=32) keeps it
// memory-bound; f32 WMMA (K=4) would not.
// ---------------------------------------------------------------------------

typedef __attribute__((ext_vector_type(16))) __bf16       v16bf;
typedef __attribute__((ext_vector_type(8)))  __bf16       bf16x8;
typedef __attribute__((ext_vector_type(8)))  float        v8f;
typedef __attribute__((ext_vector_type(4)))  float        f32x4;
typedef __attribute__((ext_vector_type(4)))  unsigned int u32x4;
typedef __attribute__((ext_vector_type(8)))  int          i32x8;
typedef __attribute__((ext_vector_type(4)))  int          i32x4;

#define TILE_M 128
#define TILE_N 64
#define TILE_K 32
// TDM pads 4 DWORDs (16B) after every 16 DWORDs (64B row) -> LDS row stride
// 80B = 40 bf16. 80 % 16 == 0 (aligned b128) and gcd(80,256)=16 -> 16 lanes
// hitting consecutive rows cover all 64 LDS banks exactly once.
#define LDT 40

union FragU { v16bf v; bf16x8 h[2]; };

// ---------------------------------------------------------------------------
// TDM issue: 2D tile load, bf16 elements, padded rows.  D# per ISA ch.8.
// 6-arg builtin form (amdgpu-toolchain / clang-23): groups 2/3 + trailing
// 8-wide group are unused for 2D tiles -> zero-filled.
// ---------------------------------------------------------------------------
__device__ __forceinline__ void tdm_load_2d(unsigned lds_off,
                                            unsigned long long gaddr,
                                            unsigned tensor_d0,   // row len (elems)
                                            unsigned tensor_d1,   // #rows in tensor
                                            unsigned tile_d0,     // tile cols (elems)
                                            unsigned tile_d1) {   // tile rows
  u32x4 g0;
  g0[0] = 1u;                                            // count=1, user mode
  g0[1] = lds_off;                                       // lds_addr (bytes)
  g0[2] = (unsigned)(gaddr & 0xffffffffu);               // global_addr[31:0]
  g0[3] = (unsigned)((gaddr >> 32) & 0x01ffffffu)        // global_addr[56:32]
          | (2u << 30);                                  // type = 2 ("image")
  i32x8 g1;
  g1[0] = (int)((1u << 16)        // data_size = 1 -> 2 bytes
                | (1u << 20)      // pad_enable
                | (3u << 22)      // pad_interval: every 16 DWORDs (64B)
                | (3u << 25));    // pad_amount: 4 DWORDs (16B)
  g1[1] = (int)(tensor_d0 << 16);                        // tensor_dim0[15:0]
  g1[2] = (int)((tensor_d0 >> 16) | (tensor_d1 << 16));  // dim0 hi | dim1 lo
  g1[3] = (int)((tensor_d1 >> 16) | (tile_d0 << 16));    // dim1 hi | tile_dim0
  g1[4] = (int)tile_d1;                                  // tile_dim1, tile_dim2=0
  g1[5] = (int)tensor_d0;                                // dim0_stride lo (row stride)
  g1[6] = 0;                                             // dim0_stride hi | dim1_stride lo
  g1[7] = 0;
  i32x4 gz4 = {0, 0, 0, 0};                              // groups 2/3 unused (2D)
  i32x8 gz8 = {0, 0, 0, 0, 0, 0, 0, 0};
  __builtin_amdgcn_tensor_load_to_lds(g0, g1, gz4, gz4, gz8, 0);
}

__device__ __forceinline__ unsigned lds_offset_of(const void* p) {
  // Flat pointers into the LDS aperture carry the LDS byte offset in the low
  // 32 bits (ISA 10.2).
  return (unsigned)(unsigned long long)(size_t)p;
}

// ---------------------------------------------------------------------------
// pass 1: x f32 -> bf16
// ---------------------------------------------------------------------------
__global__ __launch_bounds__(256)
void cvt_x_kernel(const float* __restrict__ x, __bf16* __restrict__ xb,
                  int total) {
  int i = (blockIdx.x * 256 + threadIdx.x) * 8;
  if (i >= total) return;
  f32x4 f0 = *(const f32x4*)(x + i);
  f32x4 f1 = *(const f32x4*)(x + i + 4);
  bf16x8 p;
#pragma unroll
  for (int j = 0; j < 4; ++j) {
    p[j]     = (__bf16)f0[j];
    p[j + 4] = (__bf16)f1[j];
  }
  *(bf16x8*)(xb + i) = p;
}

// ---------------------------------------------------------------------------
// pass 2: w_blocks[p,q,8,8] -> dense bf16 Wd[out][in]
// ---------------------------------------------------------------------------
__global__ __launch_bounds__(256)
void build_w_kernel(const float* __restrict__ wb, __bf16* __restrict__ wd,
                    int IN, int OUT) {
  int idx = blockIdx.x * 256 + threadIdx.x;   // one 8-float chunk per thread
  int chunks_per_row = IN >> 3;
  int o = idx / chunks_per_row;
  int i = (idx - o * chunks_per_row) * 8;
  if (o >= OUT) return;
  const float* src =
      wb + (((size_t)(o >> 3) * (size_t)(IN >> 3) + (size_t)(i >> 3)) << 6) +
      ((o & 7) << 3);
  f32x4 f0 = *(const f32x4*)(src + 0);
  f32x4 f1 = *(const f32x4*)(src + 4);
  bf16x8 p;
#pragma unroll
  for (int j = 0; j < 4; ++j) {
    p[j]     = (__bf16)f0[j];
    p[j + 4] = (__bf16)f1[j];
  }
  *(bf16x8*)(wd + (size_t)o * IN + i) = p;
}

// ---------------------------------------------------------------------------
// pass 3: bf16 GEMM with TDM double-buffered staging
// ---------------------------------------------------------------------------
__global__ __launch_bounds__(256)
void gemm_kernel(const __bf16* __restrict__ xb, const __bf16* __restrict__ wd,
                 float* __restrict__ y, int N, int IN, int OUT) {
  __shared__ __bf16 As[2][TILE_M * LDT];   // As[b][m][k], TDM-padded rows
  __shared__ __bf16 Bs[2][TILE_N * LDT];   // Bs[b][o][k]

  const int tid  = threadIdx.x;
  const int lane = tid & 31;
  const int wave = tid >> 5;            // 0..7
  const int m0   = (wave >> 1) * 32;
  const int n0   = (wave &  1) * 32;
  const int half = lane >> 4;
  const int l16  = lane & 15;

  const int tileM = blockIdx.y * TILE_M;
  const int tileN = blockIdx.x * TILE_N;
  const bool issuer = (wave == 0);

  const unsigned long long a_base =
      (unsigned long long)(size_t)(xb + (size_t)tileM * IN);
  const unsigned long long b_base =
      (unsigned long long)(size_t)(wd + (size_t)tileN * IN);

  v8f acc00 = {}, acc01 = {}, acc10 = {}, acc11 = {};

  // prologue: fill buffer 0 with the first K tile
  if (issuer) {
    tdm_load_2d(lds_offset_of(As[0]), a_base, IN, N,   TILE_K, TILE_M);
    tdm_load_2d(lds_offset_of(Bs[0]), b_base, IN, OUT, TILE_K, TILE_N);
  }

  const int ksteps = IN / TILE_K;
  int buf = 0;
  for (int ks = 0; ks < ksteps; ++ks) {
    if (issuer) {
      if (ks + 1 < ksteps) {
        // issue next buffer's tiles, then drain down to them: TDM ops from one
        // wave complete in order, so tensorcnt<=2 means the current buffer is
        // resident while the next 2 loads stay in flight.
        const unsigned long long koff =
            (unsigned long long)(ks + 1) * TILE_K * sizeof(__bf16);
        tdm_load_2d(lds_offset_of(As[buf ^ 1]), a_base + koff, IN, N,
                    TILE_K, TILE_M);
        tdm_load_2d(lds_offset_of(Bs[buf ^ 1]), b_base + koff, IN, OUT,
                    TILE_K, TILE_N);
        __builtin_amdgcn_s_wait_tensorcnt(2);
      } else {
        __builtin_amdgcn_s_wait_tensorcnt(0);
      }
    }
    __syncthreads();

    const __bf16* Ab = As[buf];
    const __bf16* Bb = Bs[buf];

    // A 16x32 bf16 frag: lane l16 holds M=l16, K runs [8h..8h+7],[8h+16..8h+23]
    // B 32x16 bf16 frag: lane l16 holds N=l16, K runs [8h..8h+7],[8h+16..8h+23]
    FragU fa0, fa1, fb0, fb1;
    {
      const __bf16* a0 = &Ab[(m0 + l16) * LDT + half * 8];
      const __bf16* a1 = &Ab[(m0 + 16 + l16) * LDT + half * 8];
      fa0.h[0] = *(const bf16x8*)(a0);
      fa0.h[1] = *(const bf16x8*)(a0 + 16);
      fa1.h[0] = *(const bf16x8*)(a1);
      fa1.h[1] = *(const bf16x8*)(a1 + 16);
      const __bf16* b0 = &Bb[(n0 + l16) * LDT + half * 8];
      const __bf16* b1 = &Bb[(n0 + 16 + l16) * LDT + half * 8];
      fb0.h[0] = *(const bf16x8*)(b0);
      fb0.h[1] = *(const bf16x8*)(b0 + 16);
      fb1.h[0] = *(const bf16x8*)(b1);
      fb1.h[1] = *(const bf16x8*)(b1 + 16);
    }

    acc00 = __builtin_amdgcn_wmma_f32_16x16x32_bf16(
        false, fa0.v, false, fb0.v, (short)0, acc00, false, false);
    acc01 = __builtin_amdgcn_wmma_f32_16x16x32_bf16(
        false, fa0.v, false, fb1.v, (short)0, acc01, false, false);
    acc10 = __builtin_amdgcn_wmma_f32_16x16x32_bf16(
        false, fa1.v, false, fb0.v, (short)0, acc10, false, false);
    acc11 = __builtin_amdgcn_wmma_f32_16x16x32_bf16(
        false, fa1.v, false, fb1.v, (short)0, acc11, false, false);

    __syncthreads();   // all waves done reading buf before it is re-filled
    buf ^= 1;
  }

  // epilogue: ReLU6 + store.  C/D layout: VGPR r, lane L -> row r+8*(L>>4),
  // col L&15 within each 16x16 tile.
  auto store_tile = [&](const v8f& a, int rbase, int cbase) {
#pragma unroll
    for (int r = 0; r < 8; ++r) {
      float v = a[r];
      v = v < 0.0f ? 0.0f : (v > 6.0f ? 6.0f : v);
      y[(size_t)(rbase + r + half * 8) * OUT + cbase + l16] = v;
    }
  };
  const int rb = tileM + m0;
  const int cb = tileN + n0;
  store_tile(acc00, rb,      cb);
  store_tile(acc01, rb,      cb + 16);
  store_tile(acc10, rb + 16, cb);
  store_tile(acc11, rb + 16, cb + 16);
}

// ---------------------------------------------------------------------------
extern "C" void kernel_launch(void* const* d_in, const int* in_sizes, int n_in,
                              void* d_out, int out_size, void* d_ws, size_t ws_size,
                              hipStream_t stream) {
  (void)n_in; (void)out_size; (void)ws_size;
  const float* x  = (const float*)d_in[0];
  const float* wb = (const float*)d_in[1];
  float* y = (float*)d_out;

  const int IN  = 1024;
  const int OUT = 1024;
  const int N   = in_sizes[0] / IN;     // 16384

  __bf16* xb = (__bf16*)d_ws;                                  // 32 MB
  __bf16* wd = (__bf16*)((char*)d_ws + (size_t)N * IN * 2);    //  2 MB

  {
    int total = N * IN;
    int blocks = (total / 8 + 255) / 256;
    cvt_x_kernel<<<blocks, 256, 0, stream>>>(x, xb, total);
  }
  {
    int blocks = ((OUT * IN / 8) + 255) / 256;
    build_w_kernel<<<blocks, 256, 0, stream>>>(wb, wd, IN, OUT);
  }
  {
    dim3 grid(OUT / TILE_N, N / TILE_M, 1);   // (16, 128)
    gemm_kernel<<<grid, 256, 0, stream>>>(xb, wd, y, N, IN, OUT);
  }
}
